// Attention_8753143349387
// MI455X (gfx1250) — matrix operand order repaired and checked by reading, not compiled, not run on hardware
//
#include <hip/hip_runtime.h>
#include <hip/hip_bf16.h>

// ---------------------------------------------------------------------------
// GQA attention for MI455X (gfx1250, wave32, WMMA).
// B=2 S=2048 D=2048 H=32 KV=8 HD=64 REP=4
// ---------------------------------------------------------------------------

#define BB 2
#define SS 2048
#define DD 2048
#define HH 32
#define KVH 8
#define HD 64
#define REP 4

typedef _Float16 v16h __attribute__((ext_vector_type(16)));
typedef float    v8f  __attribute__((ext_vector_type(8)));
typedef int      v4i  __attribute__((ext_vector_type(4)));

#ifndef __has_builtin
#define __has_builtin(x) 0
#endif

// ---- async global -> LDS staging (gfx1250 GLOBAL_LOAD_ASYNC_TO_LDS_B128) ---
#if __has_builtin(__builtin_amdgcn_global_load_async_to_lds_b128) && \
    __has_builtin(__builtin_amdgcn_s_wait_asynccnt)
__device__ __forceinline__ void async_copy_b128(void* l, const void* g) {
    __builtin_amdgcn_global_load_async_to_lds_b128((v4i*)g, (v4i*)l, 0, 0);
}
__device__ __forceinline__ void async_wait() { __builtin_amdgcn_s_wait_asynccnt(0); }
#else
__device__ __forceinline__ void async_copy_b128(void* l, const void* g) {
    *(uint4*)l = *(const uint4*)g;
}
__device__ __forceinline__ void async_wait() {}
#endif

__device__ __forceinline__ v16h ld_frag(const _Float16* p0, const _Float16* p1) {
    // Two 16B contiguous chunks -> one 32B fragment (ds_load_b128 x2 from LDS)
    v16h r;
    ((uint4*)&r)[0] = *(const uint4*)p0;
    ((uint4*)&r)[1] = *(const uint4*)p1;
    return r;
}

__device__ __forceinline__ v8f wmma_f32(v16h a, v16h b, v8f c) {
    return __builtin_amdgcn_wmma_f32_16x16x32_f16(
        /*neg_a=*/false, a, /*neg_b=*/false, b,
        /*c_mod=*/(short)0, c, /*reuse_a=*/false, /*reuse_b=*/false);
}

// ---------------------------------------------------------------------------
// f32 -> f16 elementwise convert
// ---------------------------------------------------------------------------
__global__ void cvt_f32_f16(const float* __restrict__ src, _Float16* __restrict__ dst, int n) {
    int i = blockIdx.x * 256 + threadIdx.x;
    if (i < n) dst[i] = (_Float16)src[i];
}

// ---------------------------------------------------------------------------
// C[M,N] = A[M,K] * W[N,K]^T   (A,W f16 row-major; C f32 row-major)
// WG = 256 threads = 8 waves; WG tile 128x64; wave tile 32x32; K-step 64.
// 8 WMMAs per wave per K-step on 16 ds_load_b128 (2x fragment reuse).
// ---------------------------------------------------------------------------
__global__ __launch_bounds__(256) void gemm_wmma_f16(
    const _Float16* __restrict__ A, const _Float16* __restrict__ W,
    float* __restrict__ C, int M, int N, int K)
{
    __shared__ _Float16 As[128][72];   // row stride 144B (16B aligned)
    __shared__ _Float16 Bs[64][72];

    const int t = threadIdx.x;
    const int lane = t & 31, wave = t >> 5;
    const int wrow = wave & 3, wcol = wave >> 2;       // 4x2 waves over 128x64
    const int col16 = lane & 15, half = lane >> 4;
    const int mBase = blockIdx.y * 128, nBase = blockIdx.x * 64;

    v8f acc00 = {}, acc01 = {}, acc10 = {}, acc11 = {};

    for (int k0 = 0; k0 < K; k0 += 64) {
        // stage A: 128x64 halves = 1024 x 16B, 4 per thread
        #pragma unroll
        for (int i = 0; i < 4; ++i) {
            int u = i * 256 + t;
            int row = u >> 3, col = (u & 7) * 8;
            async_copy_b128(&As[row][col], &A[(size_t)(mBase + row) * K + k0 + col]);
        }
        // stage W tile: 64x64 halves = 512 x 16B, 2 per thread
        #pragma unroll
        for (int i = 0; i < 2; ++i) {
            int u = i * 256 + t;
            int row = u >> 3, col = (u & 7) * 8;
            async_copy_b128(&Bs[row][col], &W[(size_t)(nBase + row) * K + k0 + col]);
        }
        if (k0 + 64 < K) {
            __builtin_prefetch(&A[(size_t)(mBase + (t >> 1)) * K + k0 + 64], 0, 1);
        }
        async_wait();
        __syncthreads();

        // A fragments: lane = row M=col16; K = half*16 + 0..15 contiguous
        const int am0 = wrow * 32 + col16, am1 = am0 + 16;
        v16h a00 = ld_frag(&As[am0][half * 16],      &As[am0][half * 16 + 8]);
        v16h a01 = ld_frag(&As[am0][32 + half * 16], &As[am0][32 + half * 16 + 8]);
        v16h a10 = ld_frag(&As[am1][half * 16],      &As[am1][half * 16 + 8]);
        v16h a11 = ld_frag(&As[am1][32 + half * 16], &As[am1][32 + half * 16 + 8]);
        // B fragments: lane = col N=col16; K = half*16 + 0..15 contiguous
        const int bn0 = wcol * 32 + col16, bn1 = bn0 + 16;
        v16h b00 = ld_frag(&Bs[bn0][half * 16],      &Bs[bn0][half * 16 + 8]);
        v16h b01 = ld_frag(&Bs[bn0][32 + half * 16], &Bs[bn0][32 + half * 16 + 8]);
        v16h b10 = ld_frag(&Bs[bn1][half * 16],      &Bs[bn1][half * 16 + 8]);
        v16h b11 = ld_frag(&Bs[bn1][32 + half * 16], &Bs[bn1][32 + half * 16 + 8]);

        acc00 = wmma_f32(a00, b00, acc00); acc00 = wmma_f32(a01, b01, acc00);
        acc01 = wmma_f32(a00, b10, acc01); acc01 = wmma_f32(a01, b11, acc01);
        acc10 = wmma_f32(a10, b00, acc10); acc10 = wmma_f32(a11, b01, acc10);
        acc11 = wmma_f32(a10, b10, acc11); acc11 = wmma_f32(a11, b11, acc11);
        __syncthreads();
    }

    // C layout: lane -> (M = r + 8*half, N = col16)
    const int cn = nBase + wcol * 32 + col16;
    const int cm = mBase + wrow * 32 + half * 8;
    #pragma unroll
    for (int r = 0; r < 8; ++r) {
        C[(size_t)(cm + r) * N + cn]           = acc00[r];
        C[(size_t)(cm + r) * N + cn + 16]      = acc01[r];
        C[(size_t)(cm + 16 + r) * N + cn]      = acc10[r];
        C[(size_t)(cm + 16 + r) * N + cn + 16] = acc11[r];
    }
}

// ---------------------------------------------------------------------------
// RoPE: read f32 [B,S,nh*HD], write f16 head-major [B,nh,S,HD]
// ---------------------------------------------------------------------------
__global__ void rope_kernel(const float* __restrict__ src, const float* __restrict__ cosb,
                            const float* __restrict__ sinb, _Float16* __restrict__ dst,
                            int nh, int n)
{
    int idx = blockIdx.x * 256 + threadIdx.x;
    if (idx >= n) return;
    int p = idx & (HD / 2 - 1);
    int tmp = idx >> 5;                 // HD/2 == 32
    int hh = tmp % nh; tmp /= nh;
    int s = tmp % SS;  int bb = tmp / SS;
    int ld = nh * HD;
    float c  = cosb[s * (HD / 2) + p];
    float sn = sinb[s * (HD / 2) + p];
    size_t si = ((size_t)bb * SS + s) * ld + hh * HD + 2 * p;
    float xr = src[si], xi = src[si + 1];
    size_t di = (((size_t)bb * nh + hh) * SS + s) * HD + 2 * p;
    dst[di]     = (_Float16)(xr * c - xi * sn);
    dst[di + 1] = (_Float16)(xr * sn + xi * c);
}

// V: f32 [B,S,KV*HD] -> f16 [B,KV,S,HD]
__global__ void cvt_v_kernel(const float* __restrict__ src, _Float16* __restrict__ dst, int n) {
    int idx = blockIdx.x * 256 + threadIdx.x;
    if (idx >= n) return;
    int d = idx & (HD - 1);
    int tmp = idx >> 6;
    int hh = tmp % KVH; tmp /= KVH;
    int s = tmp % SS;   int bb = tmp / SS;
    dst[(((size_t)bb * KVH + hh) * SS + s) * HD + d] =
        (_Float16)src[((size_t)bb * SS + s) * (KVH * HD) + hh * HD + d];
}

// ---------------------------------------------------------------------------
// Flash attention: grid = B*H*(S/64), block = 128 (4 waves).
// Each wave owns 16 q rows; 32 keys per iteration; online softmax in C-layout.
// ---------------------------------------------------------------------------
__global__ __launch_bounds__(128) void flash_attn(
    const _Float16* __restrict__ Q,   // [B,H,S,HD]
    const _Float16* __restrict__ Kt,  // [B,KV,S,HD]
    const _Float16* __restrict__ Vt,  // [B,KV,S,HD]
    _Float16* __restrict__ O)         // [B,S,D]
{
    __shared__ _Float16 Qs[64][72];      // q rows x hd
    __shared__ _Float16 Ks[32][72];      // key rows x hd
    __shared__ _Float16 Vs[64][40];      // hd x key (transposed)
    __shared__ _Float16 Ps[4][16][40];   // per-wave P tile 16 x 32

    const int t = threadIdx.x, lane = t & 31, wave = t >> 5;
    const int col16 = lane & 15, half = lane >> 4;

    const int nqt = SS / 64;
    const int qt = blockIdx.x % nqt;
    const int h  = (blockIdx.x / nqt) % HH;
    const int b  = blockIdx.x / (nqt * HH);
    const int kvh = h / REP;
    const int qBase = qt * 64;

    const _Float16* Qp = Q  + (((size_t)b * HH  + h)   * SS + qBase) * HD;
    const _Float16* Kp = Kt + (((size_t)b * KVH + kvh) * SS) * HD;
    const _Float16* Vp = Vt + (((size_t)b * KVH + kvh) * SS) * HD;

    // stage Q 64x64 (async global -> LDS)
    #pragma unroll
    for (int i = 0; i < 4; ++i) {
        int e = (i * 128 + t) * 8, row = e >> 6, col = e & 63;
        async_copy_b128(&Qs[row][col], &Qp[row * HD + col]);
    }
    async_wait();
    __syncthreads();

    const int am = wave * 16 + col16;
    v16h qa0 = ld_frag(&Qs[am][half * 16],      &Qs[am][half * 16 + 8]);
    v16h qa1 = ld_frag(&Qs[am][32 + half * 16], &Qs[am][32 + half * 16 + 8]);

    float m[8], l[8];
    v8f o0 = {}, o1 = {}, o2 = {}, o3 = {};
    #pragma unroll
    for (int r = 0; r < 8; ++r) { m[r] = -1e30f; l[r] = 0.f; }

    const int qRowBase = qBase + wave * 16 + half * 8;    // + r gives this lane's row
    const int nkt = (qBase + 64) / 32;                    // causal: keys <= last q row

    for (int j = 0; j < nkt; ++j) {
        const int kBase = j * 32;
        // stage K 32x64 (async; B-fragments for QK^T read contiguous hd)
        #pragma unroll
        for (int i = 0; i < 2; ++i) {
            int e = (i * 128 + t) * 8, row = e >> 6, col = e & 63;
            async_copy_b128(&Ks[row][col], &Kp[(size_t)(kBase + row) * HD + col]);
        }
        // stage V transposed -> Vs[hd][key] (B-fragments for PV read contiguous keys)
        #pragma unroll
        for (int i = 0; i < 2; ++i) {
            int e = (i * 128 + t) * 8, row = e >> 6, col = e & 63;
            uint4 vv = *(const uint4*)&Vp[(size_t)(kBase + row) * HD + col];
            const _Float16* vh = (const _Float16*)&vv;
            #pragma unroll
            for (int u = 0; u < 8; ++u) Vs[col + u][row] = vh[u];
        }
        async_wait();
        __syncthreads();

        // ---- scores: two 16-key tiles -------------------------------------
        v8f s0 = {}, s1 = {};
        {
            v16h kb = ld_frag(&Ks[col16][half * 16], &Ks[col16][half * 16 + 8]);
            s0 = wmma_f32(qa0, kb, s0);
            kb = ld_frag(&Ks[col16][32 + half * 16], &Ks[col16][32 + half * 16 + 8]);
            s0 = wmma_f32(qa1, kb, s0);
        }
        {
            v16h kb = ld_frag(&Ks[16 + col16][half * 16], &Ks[16 + col16][half * 16 + 8]);
            s1 = wmma_f32(qa0, kb, s1);
            kb = ld_frag(&Ks[16 + col16][32 + half * 16], &Ks[16 + col16][32 + half * 16 + 8]);
            s1 = wmma_f32(qa1, kb, s1);
        }

        // ---- causal mask + scale + online softmax -------------------------
        float mt[8];
        #pragma unroll
        for (int r = 0; r < 8; ++r) {
            int q  = qRowBase + r;
            int k0 = kBase + col16, k1 = kBase + 16 + col16;
            float a  = (k0 <= q) ? s0[r] * 0.125f : -1e30f;
            float bb = (k1 <= q) ? s1[r] * 0.125f : -1e30f;
            s0[r] = a; s1[r] = bb;
            mt[r] = fmaxf(a, bb);
        }
        #pragma unroll
        for (int msk = 1; msk < 16; msk <<= 1)
            #pragma unroll
            for (int r = 0; r < 8; ++r)
                mt[r] = fmaxf(mt[r], __shfl_xor(mt[r], msk, 32));

        float p0[8], p1[8], rs[8];
        #pragma unroll
        for (int r = 0; r < 8; ++r) {
            float mn = fmaxf(m[r], mt[r]);
            float sc = __expf(m[r] - mn);
            m[r] = mn; l[r] *= sc;
            o0[r] *= sc; o1[r] *= sc; o2[r] *= sc; o3[r] *= sc;
            p0[r] = __expf(s0[r] - mn);
            p1[r] = __expf(s1[r] - mn);
            rs[r] = p0[r] + p1[r];
        }
        #pragma unroll
        for (int msk = 1; msk < 16; msk <<= 1)
            #pragma unroll
            for (int r = 0; r < 8; ++r)
                rs[r] += __shfl_xor(rs[r], msk, 32);
        #pragma unroll
        for (int r = 0; r < 8; ++r) l[r] += rs[r];

        // ---- P: C-layout -> A-fragment via per-wave LDS tile --------------
        #pragma unroll
        for (int r = 0; r < 8; ++r) {
            Ps[wave][r + half * 8][col16]      = (_Float16)p0[r];
            Ps[wave][r + half * 8][16 + col16] = (_Float16)p1[r];
        }
        v16h pa = ld_frag(&Ps[wave][col16][half * 16], &Ps[wave][col16][half * 16 + 8]);

        // ---- O += P * V ---------------------------------------------------
        {
            v16h vb = ld_frag(&Vs[col16][half * 16], &Vs[col16][half * 16 + 8]);
            o0 = wmma_f32(pa, vb, o0);
            vb = ld_frag(&Vs[16 + col16][half * 16], &Vs[16 + col16][half * 16 + 8]);
            o1 = wmma_f32(pa, vb, o1);
            vb = ld_frag(&Vs[32 + col16][half * 16], &Vs[32 + col16][half * 16 + 8]);
            o2 = wmma_f32(pa, vb, o2);
            vb = ld_frag(&Vs[48 + col16][half * 16], &Vs[48 + col16][half * 16 + 8]);
            o3 = wmma_f32(pa, vb, o3);
        }
        __syncthreads();
    }

    // ---- epilogue: O / l -> f16 [B,S,D] -----------------------------------
    float inv[8];
    #pragma unroll
    for (int r = 0; r < 8; ++r) inv[r] = 1.0f / l[r];
    #pragma unroll
    for (int r = 0; r < 8; ++r) {
        size_t base = ((size_t)b * SS + (qRowBase + r)) * DD + h * HD + col16;
        O[base]      = (_Float16)(o0[r] * inv[r]);
        O[base + 16] = (_Float16)(o1[r] * inv[r]);
        O[base + 32] = (_Float16)(o2[r] * inv[r]);
        O[base + 48] = (_Float16)(o3[r] * inv[r]);
    }
}

// ---------------------------------------------------------------------------
// Host launcher
// ---------------------------------------------------------------------------
extern "C" void kernel_launch(void* const* d_in, const int* in_sizes, int n_in,
                              void* d_out, int out_size, void* d_ws, size_t ws_size,
                              hipStream_t stream)
{
    const float* x    = (const float*)d_in[0];
    const float* fcos = (const float*)d_in[1];
    const float* fsin = (const float*)d_in[2];
    const float* wq   = (const float*)d_in[3];
    const float* wk   = (const float*)d_in[4];
    const float* wv   = (const float*)d_in[5];
    const float* wo   = (const float*)d_in[6];

    const size_t MS = (size_t)BB * SS;          // 4096 rows
    char* w = (char*)d_ws;
    size_t off = 0;
    auto alloc = [&](size_t bytes) { size_t o = off; off += (bytes + 255) & ~(size_t)255; return o; };

    _Float16* xh  = (_Float16*)(w + alloc(MS * DD * 2));
    _Float16* wqh = (_Float16*)(w + alloc((size_t)DD * DD * 2));
    _Float16* wkh = (_Float16*)(w + alloc((size_t)KVH * HD * DD * 2));
    _Float16* wvh = (_Float16*)(w + alloc((size_t)KVH * HD * DD * 2));
    _Float16* woh = (_Float16*)(w + alloc((size_t)DD * DD * 2));
    float*    Qf  = (float*)   (w + alloc(MS * DD * 4));
    float*    Kf  = (float*)   (w + alloc(MS * KVH * HD * 4));
    float*    Vf  = (float*)   (w + alloc(MS * KVH * HD * 4));
    _Float16* Qh  = (_Float16*)(w + alloc(MS * DD * 2));
    _Float16* Kh  = (_Float16*)(w + alloc(MS * KVH * HD * 2));
    _Float16* Vh  = (_Float16*)(w + alloc(MS * KVH * HD * 2));
    _Float16* Oh  = (_Float16*)(w + alloc(MS * DD * 2));

    auto cvt = [&](const float* s, _Float16* d, int n) {
        cvt_f32_f16<<<(n + 255) / 256, 256, 0, stream>>>(s, d, n);
    };
    cvt(x,  xh,  (int)(MS * DD));
    cvt(wq, wqh, DD * DD);
    cvt(wk, wkh, KVH * HD * DD);
    cvt(wv, wvh, KVH * HD * DD);
    cvt(wo, woh, DD * DD);

    // projections: C = X * W^T   (WG tile 128x64)
    gemm_wmma_f16<<<dim3(DD / 64, MS / 128), 256, 0, stream>>>(xh, wqh, Qf, (int)MS, DD, DD);
    gemm_wmma_f16<<<dim3((KVH * HD) / 64, MS / 128), 256, 0, stream>>>(xh, wkh, Kf, (int)MS, KVH * HD, DD);
    gemm_wmma_f16<<<dim3((KVH * HD) / 64, MS / 128), 256, 0, stream>>>(xh, wvh, Vf, (int)MS, KVH * HD, DD);

    // RoPE + head-major layouts
    {
        int nq = BB * SS * HH * (HD / 2);
        rope_kernel<<<(nq + 255) / 256, 256, 0, stream>>>(Qf, fcos, fsin, Qh, HH, nq);
        int nk = BB * SS * KVH * (HD / 2);
        rope_kernel<<<(nk + 255) / 256, 256, 0, stream>>>(Kf, fcos, fsin, Kh, KVH, nk);
        int nv = BB * SS * KVH * HD;
        cvt_v_kernel<<<(nv + 255) / 256, 256, 0, stream>>>(Vf, Vh, nv);
    }

    // flash attention
    flash_attn<<<BB * HH * (SS / 64), 128, 0, stream>>>(Qh, Kh, Vh, Oh);

    // output projection -> f32 d_out
    gemm_wmma_f16<<<dim3(DD / 64, MS / 128), 256, 0, stream>>>(Oh, woh, (float*)d_out, (int)MS, DD, DD);
}
